// CrossAttention_6511170421543
// MI455X (gfx1250) — matrix operand order repaired and checked by reading, not compiled, not run on hardware
//
#include <hip/hip_runtime.h>
#include <hip/hip_bf16.h>
#include <stdint.h>

// ---------------- problem constants (from setup_inputs) ----------------
static constexpr int BATCH = 8;
static constexpr int CQ    = 192;        // q channels
static constexpr int CKV   = 384;        // kv channels (k=192 + v=192)
static constexpr int NHEAD = 4;
static constexpr int DHEAD = 48;         // CQ / NHEAD
static constexpr int SDIM  = 128 * 128;  // spatial = 16384

typedef __attribute__((ext_vector_type(16))) __bf16 v16bf;
typedef __attribute__((ext_vector_type(8)))  float  v8f;

union Frag16 { uint4 q[2]; v16bf v; };   // 32 bytes = 16 bf16

static __device__ __forceinline__ unsigned short f2bf(float f) {
    unsigned int u = __float_as_uint(f);
    u += 0x7FFFu + ((u >> 16) & 1u);     // round-to-nearest-even
    return (unsigned short)(u >> 16);
}
static __device__ __forceinline__ v8f vzero8() {
    v8f z;
    #pragma unroll
    for (int i = 0; i < 8; ++i) z[i] = 0.0f;
    return z;
}

// Async global->LDS b128 copy (CDNA5, tracked by ASYNCcnt).
// LDS destination: low 32 bits of the generic pointer == LDS byte offset
// (aperture rule).
static __device__ __forceinline__ void async_copy_b128(void* lds_dst, const void* gsrc) {
    unsigned       ldsoff = (unsigned)(uintptr_t)lds_dst;
    unsigned long long ga = (unsigned long long)(uintptr_t)gsrc;
    asm volatile("global_load_async_to_lds_b128 %0, %1, off"
                 :: "v"(ldsoff), "v"(ga) : "memory");
}
static __device__ __forceinline__ void wait_asynccnt0() {
    asm volatile("s_wait_asynccnt 0" ::: "memory");
}

// ---------------- fp32 -> bf16 weight conversion ----------------
__global__ void f32_to_bf16_kernel(const float* __restrict__ src,
                                   unsigned short* __restrict__ dst, int n) {
    int i = blockIdx.x * 256 + threadIdx.x;
    if (i < n) dst[i] = f2bf(src[i]);
}

// ---------------- 1x1-conv channel GEMM via WMMA bf16 ----------------
// Out[b][m][s] = sum_ch A[m][ch] * X[b][ch][s]
// A: bf16 [M x K] (optionally per-batch via aBatchStride elements)
// X: fp32 or bf16 [b][K][SDIM] (xBatchStride elements per batch)
// Out: bf16 or fp32 [b][M][SDIM]
// Optional per-output-channel sum-of-squares accumulation (for L2 norms).
template<bool XF32, bool OUTF32, bool SUMSQ>
__global__ __launch_bounds__(256)
void conv1x1_wmma_kernel(const unsigned short* __restrict__ A,
                         const void* __restrict__ X,
                         void* __restrict__ Out,
                         float* __restrict__ sumsq,
                         int M, int K,
                         long aBatchStride, long xBatchStride,
                         int ssqStride, int ssqLimit)
{
    constexpr int BM = 64, BN = 128, BK = 32;
    __shared__ __align__(16) unsigned short ldsA[BM * BK];   // [m][k] row-major
    __shared__ __align__(16) unsigned short ldsBt[BN * BK];  // [n][k] (transposed)

    const int tid   = threadIdx.x;
    const int lane  = tid & 31;
    const int wave  = tid >> 5;          // 0..7
    const int waveM = wave >> 1;         // 0..3 (16-row strips of BM=64)
    const int waveN = wave & 1;          // 0..1 (64-col strips of BN=128)
    const int b     = blockIdx.z;
    const int mtile = blockIdx.y;
    const int s0    = blockIdx.x * BN;

    const unsigned short* Ab = A + (size_t)b * aBatchStride + (size_t)mtile * BM * K;

    v8f acc[4];
    #pragma unroll
    for (int t = 0; t < 4; ++t) acc[t] = vzero8();

    for (int k0 = 0; k0 < K; k0 += BK) {
        // Prefetch next K-slab of X into cache (global_prefetch_b8)
        if (k0 + BK < K) {
            int ch = tid >> 5;                  // 0..7 (subset hint)
            int sp = (tid & 31) * 4;
            if (XF32)
                __builtin_prefetch((const float*)X + (size_t)b * xBatchStride +
                                   (size_t)(k0 + BK + ch) * SDIM + s0 + sp, 0, 1);
            else
                __builtin_prefetch((const unsigned short*)X + (size_t)b * xBatchStride +
                                   (size_t)(k0 + BK + ch) * SDIM + s0 + sp, 0, 1);
        }
        __syncthreads();   // previous iteration's LDS reads complete
        // --- stage A tile: 64x32 bf16, one async b128 per thread (ASYNCcnt) ---
        {
            int r = tid >> 2;                   // 0..63
            int c = (tid & 3) * 8;              // 0,8,16,24
            async_copy_b128(&ldsA[r * BK + c], Ab + (size_t)r * K + k0 + c);
        }
        // --- stage B tile transposed: ldsBt[s][ch], converting fp32->bf16 on the fly
        //     (overlaps with the in-flight async weight-tile copy) ---
        #pragma unroll
        for (int j = 0; j < 4; ++j) {
            int idx = j * 256 + tid;            // 0..1023
            int ch  = idx >> 5;                 // 0..31
            int sp  = (idx & 31) * 4;           // 0..124
            if (XF32) {
                const float* xs = (const float*)X + (size_t)b * xBatchStride +
                                  (size_t)(k0 + ch) * SDIM + s0 + sp;
                float4 f = *(const float4*)xs;
                ldsBt[(sp + 0) * BK + ch] = f2bf(f.x);
                ldsBt[(sp + 1) * BK + ch] = f2bf(f.y);
                ldsBt[(sp + 2) * BK + ch] = f2bf(f.z);
                ldsBt[(sp + 3) * BK + ch] = f2bf(f.w);
            } else {
                const unsigned short* xs = (const unsigned short*)X + (size_t)b * xBatchStride +
                                           (size_t)(k0 + ch) * SDIM + s0 + sp;
                uint2 h4 = *(const uint2*)xs;   // 4 bf16
                ldsBt[(sp + 0) * BK + ch] = (unsigned short)(h4.x);
                ldsBt[(sp + 1) * BK + ch] = (unsigned short)(h4.x >> 16);
                ldsBt[(sp + 2) * BK + ch] = (unsigned short)(h4.y);
                ldsBt[(sp + 3) * BK + ch] = (unsigned short)(h4.y >> 16);
            }
        }
        wait_asynccnt0();   // this wave's async LDS writes done
        __syncthreads();    // ... and every other wave's too
        // --- A fragment (16x32): lane<16 needs K {0..7,16..23}, lane>=16 {8..15,24..31}
        Frag16 fa;
        {
            int row = waveM * 16 + (lane & 15);
            int kb  = (lane < 16) ? 0 : 8;
            fa.q[0] = *(const uint4*)&ldsA[row * BK + kb];
            fa.q[1] = *(const uint4*)&ldsA[row * BK + kb + 16];
        }
        #pragma unroll
        for (int t = 0; t < 4; ++t) {
            // B fragment (32x16): lane<16 K {0..7,16..23}, lane>=16 K {8..15,24..31}
            Frag16 fb;
            int n  = waveN * 64 + t * 16 + (lane & 15);
            int kb = (lane < 16) ? 0 : 8;
            fb.q[0] = *(const uint4*)&ldsBt[n * BK + kb];
            fb.q[1] = *(const uint4*)&ldsBt[n * BK + kb + 16];
            acc[t] = __builtin_amdgcn_wmma_f32_16x16x32_bf16(
                false, fa.v, false, fb.v, (short)0, acc[t], false, false);
        }
    }

    // --- store + optional per-channel sum-of-squares ---
    #pragma unroll
    for (int t = 0; t < 4; ++t) {
        int nbase = s0 + waveN * 64 + t * 16 + (lane & 15);
        #pragma unroll
        for (int r = 0; r < 8; ++r) {
            int mrow = mtile * BM + waveM * 16 + r + ((lane >> 4) << 3);
            float v = acc[t][r];
            if (OUTF32)
                ((float*)Out)[((size_t)b * M + mrow) * SDIM + nbase] = v;
            else
                ((unsigned short*)Out)[((size_t)b * M + mrow) * SDIM + nbase] = f2bf(v);
            if (SUMSQ) {
                float ss = v * v;                 // reduce across 16 lanes sharing mrow
                ss += __shfl_xor(ss, 1);
                ss += __shfl_xor(ss, 2);
                ss += __shfl_xor(ss, 4);
                ss += __shfl_xor(ss, 8);
                if ((lane & 15) == 0 && mrow < ssqLimit)
                    atomicAdd(&sumsq[(size_t)b * ssqStride + mrow], ss);
            }
        }
    }
}

// ---------------- Gram matrix G = q . k^T over s (split-K, fp32 atomics) -------
// q: bf16 [b][192][S]  (head n -> channels n*48..n*48+47)
// k: bf16 [b][384][S]  (k part = channels 0..191)
// G: fp32 [b][nh][48][48], pre-zeroed
__global__ __launch_bounds__(288)
void gram_wmma_kernel(const unsigned short* __restrict__ qb,
                      const unsigned short* __restrict__ kb,
                      float* __restrict__ G, int chunk)
{
    const int lane = threadIdx.x & 31;
    const int wave = threadIdx.x >> 5;   // 0..8 : 3x3 WMMA tiles of 48x48
    const int mt = wave / 3, nt = wave % 3;
    const int bn = blockIdx.x;           // b*NHEAD + n
    const int b = bn >> 2, n = bn & 3;

    const unsigned short* qrow = qb + ((size_t)b * CQ  + n * DHEAD + mt * 16 + (lane & 15)) * SDIM;
    const unsigned short* krow = kb + ((size_t)b * CKV + n * DHEAD + nt * 16 + (lane & 15)) * SDIM;
    const int aoff = (lane < 16) ? 0 : 8;    // A 16-bit layout K-offset
    const int boff = (lane < 16) ? 0 : 8;    // B 16-bit layout K-offset
    const int sbeg = blockIdx.y * chunk;

    v8f acc = vzero8();
    for (int s0 = sbeg; s0 < sbeg + chunk; s0 += 32) {
        Frag16 fa, fb;
        fa.q[0] = *(const uint4*)(qrow + s0 + aoff);
        fa.q[1] = *(const uint4*)(qrow + s0 + aoff + 16);
        fb.q[0] = *(const uint4*)(krow + s0 + boff);
        fb.q[1] = *(const uint4*)(krow + s0 + boff + 16);
        acc = __builtin_amdgcn_wmma_f32_16x16x32_bf16(
            false, fa.v, false, fb.v, (short)0, acc, false, false);
    }
    float* Gp = G + (size_t)bn * DHEAD * DHEAD;
    int e = nt * 16 + (lane & 15);
    #pragma unroll
    for (int r = 0; r < 8; ++r) {
        int d = mt * 16 + r + ((lane >> 4) << 3);
        atomicAdd(&Gp[d * DHEAD + e], acc[r]);
    }
}

// ---------------- normalize + softmax + fold attn into Wp ----------------
// Mfold[b][o][n*48+e] = sum_d Wp[o][n*48+d] * softmax_e(G[d][e]*temp[n]/(|q_d||k_e|))
__global__ __launch_bounds__(256)
void attn_fold_kernel(const float* __restrict__ G,
                      const float* __restrict__ ssq_q,
                      const float* __restrict__ ssq_k,
                      const float* __restrict__ Wp,
                      const float* __restrict__ temp,
                      unsigned short* __restrict__ Mfold)
{
    __shared__ float sA[DHEAD * DHEAD];
    __shared__ float nq[DHEAD], nk[DHEAD];
    const int bn = blockIdx.x, b = bn >> 2, n = bn & 3;
    const int tid = threadIdx.x;

    if (tid < DHEAD) {
        nq[tid] = fmaxf(sqrtf(ssq_q[b * CQ  + n * DHEAD + tid]), 1e-12f);
        nk[tid] = fmaxf(sqrtf(ssq_k[b * CKV + n * DHEAD + tid]), 1e-12f);
    }
    __syncthreads();
    const float ts = temp[n];
    const float* Gp = G + (size_t)bn * DHEAD * DHEAD;
    for (int i = tid; i < DHEAD * DHEAD; i += 256) {
        int d = i / DHEAD, e = i - d * DHEAD;
        sA[i] = Gp[i] * ts / (nq[d] * nk[e]);
    }
    __syncthreads();
    if (tid < DHEAD) {                       // row-wise softmax, one thread per row
        float* row = &sA[tid * DHEAD];
        float m = row[0];
        for (int e = 1; e < DHEAD; ++e) m = fmaxf(m, row[e]);
        float sum = 0.0f;
        for (int e = 0; e < DHEAD; ++e) { float ex = __expf(row[e] - m); row[e] = ex; sum += ex; }
        float inv = 1.0f / sum;
        for (int e = 0; e < DHEAD; ++e) row[e] *= inv;
    }
    __syncthreads();
    for (int i = tid; i < CQ * DHEAD; i += 256) {
        int o = i / DHEAD, e = i - o * DHEAD;
        const float* wr = Wp + o * CQ + n * DHEAD;
        float s = 0.0f;
        #pragma unroll 8
        for (int d = 0; d < DHEAD; ++d) s = fmaf(wr[d], sA[d * DHEAD + e], s);
        Mfold[((size_t)b * CQ + o) * CQ + n * DHEAD + e] = f2bf(s);
    }
}

// ---------------- launcher ----------------
extern "C" void kernel_launch(void* const* d_in, const int* in_sizes, int n_in,
                              void* d_out, int out_size, void* d_ws, size_t ws_size,
                              hipStream_t stream) {
    const float* x    = (const float*)d_in[0];
    const float* y    = (const float*)d_in[1];
    const float* Wq   = (const float*)d_in[2];
    const float* Wkv  = (const float*)d_in[3];
    const float* Wp   = (const float*)d_in[4];
    const float* temp = (const float*)d_in[5];

    // workspace layout (bytes)
    char* ws = (char*)d_ws;
    unsigned short* qbuf  = (unsigned short*)(ws);                         //  50,331,648
    unsigned short* kvbuf = (unsigned short*)(ws + 50331648);              // 100,663,296
    unsigned short* Wq_b  = (unsigned short*)(ws + 150994944);             //      73,728
    unsigned short* Wkv_b = (unsigned short*)(ws + 151068672);             //     294,912
    unsigned short* Mfold = (unsigned short*)(ws + 151363584);             //     589,824
    float*          G     = (float*)(ws + 151953408);                      //     294,912
    float*          ssq_q = (float*)(ws + 152248320);                      //       6,144
    float*          ssq_k = (float*)(ws + 152254464);                      //      12,288
    // zero accumulation buffers (G + ssq_q + ssq_k are contiguous)
    hipMemsetAsync(G, 0, 294912 + 6144 + 12288, stream);

    f32_to_bf16_kernel<<<(CQ * CQ + 255) / 256, 256, 0, stream>>>(Wq, Wq_b, CQ * CQ);
    f32_to_bf16_kernel<<<(CKV * CKV + 255) / 256, 256, 0, stream>>>(Wkv, Wkv_b, CKV * CKV);

    // q = Wq @ x  (+ sum q^2 per channel)
    dim3 g1(SDIM / 128, CQ / 64, BATCH);
    conv1x1_wmma_kernel<true, false, true><<<g1, 256, 0, stream>>>(
        Wq_b, x, qbuf, ssq_q, CQ, CQ, 0, (long)CQ * SDIM, CQ, CQ);
    // kv = Wkv @ y  (+ sum k^2 for channels < 192)
    dim3 g2(SDIM / 128, CKV / 64, BATCH);
    conv1x1_wmma_kernel<true, false, true><<<g2, 256, 0, stream>>>(
        Wkv_b, y, kvbuf, ssq_k, CKV, CKV, 0, (long)CKV * SDIM, CKV, CQ);

    // G = q . k^T  (split-K over 16 chunks of 1024)
    dim3 g3(BATCH * NHEAD, 16);
    gram_wmma_kernel<<<g3, 288, 0, stream>>>(qbuf, kvbuf, G, SDIM / 16);

    // softmax + fold into Wp
    attn_fold_kernel<<<BATCH * NHEAD, 256, 0, stream>>>(G, ssq_q, ssq_k, Wp, temp, Mfold);

    // out = Mfold[b] @ v   (v = kv channels 192..383)
    dim3 g4(SDIM / 128, CQ / 64, BATCH);
    conv1x1_wmma_kernel<false, true, false><<<g4, 256, 0, stream>>>(
        Mfold, kvbuf + (size_t)CQ * SDIM, (float*)d_out, nullptr,
        CQ, CQ, (long)CQ * CQ, (long)CKV * SDIM, 0, 0);
}
